// GCN_12567074308294
// MI455X (gfx1250) — matrix operand order repaired and checked by reading, not compiled, not run on hardware
//
#include <hip/hip_runtime.h>
#include <math.h>
#include <stdint.h>

// Problem constants (match reference)
#define NNODES  50000
#define NEDGES  800000
#define IN_DIM  128
#define HDIM    96
#define CDIM    10
#define LAYERS  4
#define KEIG    16
#define LN_EPS  1e-5f

typedef __attribute__((ext_vector_type(16))) _Float16 v16h;
typedef __attribute__((ext_vector_type(8)))  _Float16 v8h;
typedef __attribute__((ext_vector_type(8)))  float    v8f;

__device__ __forceinline__ float gelu_exact(float x) {
  return 0.5f * x * (1.0f + erff(x * 0.70710678118654752440f));
}
__device__ __forceinline__ float fixnum(float x) {  // jnp.nan_to_num
  const float m = 3.4028234663852886e38f;
  if (x != x) return 0.f;
  if (x >  m) return m;
  if (x < -m) return -m;
  return x;
}
__device__ __forceinline__ void atomAddF(float* p, float v) { unsafeAtomicAdd(p, v); }

// fp32 -> fp16 mirror of an activation matrix (A operands of the WMMA GEMMs)
__global__ void tof16_kernel(const float* __restrict__ in,
                             _Float16* __restrict__ out, int n)
{
  int i = blockIdx.x * blockDim.x + threadIdx.x;
  if (i < n) out[i] = (_Float16)in[i];
}

// ---------------------------------------------------------------------------
// Generic WMMA GEMM: out[M,96] = act(A16[M,K] @ B[K,96] (+ bias))
// K in {96,128}. Block = 128 threads (4 waves); each wave owns 16 rows,
// block owns 64 rows.
//  - A panel (already f16 in global) is staged to LDS with
//    GLOBAL_LOAD_ASYNC_TO_LDS_B128 (ASYNCcnt), bypassing the VGPR file.
//  - B (f32 weights) is converted to f16 *while* the async copies fly,
//    stored transposed (+8-half pad) so every WMMA fragment is two aligned
//    contiguous ds_load_b128.
// act: 0 = none, 1 = relu, 2 = gelu
// ---------------------------------------------------------------------------
__global__ __launch_bounds__(128) void gemm_wmma_kernel(
    const _Float16* __restrict__ A, const float* __restrict__ B,
    const float* __restrict__ bias, float* __restrict__ out,
    int M, int K, int has_bias, int act)
{
  __shared__ __align__(16) _Float16 Ah[64][IN_DIM + 8];   // [row][k]
  __shared__ __align__(16) _Float16 Bt[HDIM][IN_DIM + 8]; // [n][k] (transposed)

  const int tid  = threadIdx.x;
  const int row0 = blockIdx.x * 64;

  // Kick off async global->LDS copies of the A panel (16B per lane-op).
  // Row stride in LDS is 272B (16B-aligned); chunk offsets are 16B multiples.
  const int chunksPerRow = K / 8;   // 16-byte chunks per row
  for (int c = tid; c < 64 * chunksPerRow; c += 128) {
    int r = c / chunksPerRow, j = c % chunksPerRow;
    int gr = row0 + r;
    uint32_t ldsa = (uint32_t)(uintptr_t)&Ah[r][8 * j];
    if (gr < M) {
      unsigned long long gaddr =
          (unsigned long long)(uintptr_t)(A + (long long)gr * K + 8 * j);
      asm volatile("global_load_async_to_lds_b128 %0, %1, off"
                   :: "v"(ldsa), "v"(gaddr) : "memory");
    } else {
      v8h z = {};
      *(v8h*)&Ah[r][8 * j] = z;   // zero-fill tail rows
    }
  }

  // Meanwhile: stage B (K x 96, row-major f32) transposed into LDS as f16
  for (int i = tid; i < K * HDIM; i += 128) {
    int k = i / HDIM, n = i % HDIM;
    Bt[n][k] = (_Float16)B[i];
  }

  asm volatile("s_wait_asynccnt 0x0" ::: "memory");  // my async copies done
  __syncthreads();                                   // everyone's done

  const int wave = tid >> 5;
  const int lane = tid & 31;
  const int hi   = lane >> 4;   // 0: lanes 0-15, 1: lanes 16-31
  const int l15  = lane & 15;
  const int ar   = 16 * wave + l15;  // A-fragment row within block

  v8f acc[6] = {};  // 6 column tiles of 16 -> 96 cols

  for (int kc = 0; kc < K; kc += 32) {
    // A fragment: halves 0..7 = K(kc+8*hi .. +7), halves 8..15 = +16
    v8h alo = *(const v8h*)&Ah[ar][kc + 8 * hi];
    v8h ahi = *(const v8h*)&Ah[ar][kc + 16 + 8 * hi];
    v16h afrag = __builtin_shufflevector(alo, ahi,
        0,1,2,3,4,5,6,7,8,9,10,11,12,13,14,15);
#pragma unroll
    for (int t = 0; t < 6; ++t) {
      // B fragment: lane holds col N=16t+l15, halves 0..7 = K(kc+8*hi .. +7), halves 8..15 = +16
      v8h blo = *(const v8h*)&Bt[16 * t + l15][kc + 8 * hi];
      v8h bhi = *(const v8h*)&Bt[16 * t + l15][kc + 16 + 8 * hi];
      v16h bfrag = __builtin_shufflevector(blo, bhi,
          0,1,2,3,4,5,6,7,8,9,10,11,12,13,14,15);
      acc[t] = __builtin_amdgcn_wmma_f32_16x16x32_f16(
          false, afrag, false, bfrag, (short)0, acc[t], false, false);
    }
  }

  // Epilogue: lane L, VGPR r holds D[M = r + 8*hi][N = l15] per tile
#pragma unroll
  for (int r = 0; r < 8; ++r) {
    int grow = row0 + 16 * wave + 8 * hi + r;
    if (grow < M) {
#pragma unroll
      for (int t = 0; t < 6; ++t) {
        int col = 16 * t + l15;
        float v = acc[t][r];
        if (has_bias) v += bias[col];
        if (act == 1)      v = fmaxf(v, 0.0f);
        else if (act == 2) v = gelu_exact(v);
        out[(long long)grow * HDIM + col] = v;
      }
    }
  }
}

// ---------------------------------------------------------------------------
// Row-wise LayerNorm over 96 features. 8 waves/block, one row per wave32,
// 3 features per lane. Optional elementwise GELU before the norm and
// optional accumulate (out += result) for the residual stream.
// ---------------------------------------------------------------------------
__global__ __launch_bounds__(256) void ln_kernel(
    const float* __restrict__ in, const float* __restrict__ g,
    const float* __restrict__ b, float* __restrict__ out,
    int nrows, int do_gelu, int acc)
{
  int wave = threadIdx.x >> 5;
  int lane = threadIdx.x & 31;
  int row  = blockIdx.x * 8 + wave;
  if (row >= nrows) return;

  const float* rp = in + (long long)row * HDIM;
  float v[3];
#pragma unroll
  for (int j = 0; j < 3; ++j) {
    float x = rp[lane + 32 * j];
    v[j] = do_gelu ? gelu_exact(x) : x;
  }
  float s  = v[0] + v[1] + v[2];
  float s2 = v[0]*v[0] + v[1]*v[1] + v[2]*v[2];
#pragma unroll
  for (int off = 16; off >= 1; off >>= 1) {
    s  += __shfl_xor(s,  off, 32);
    s2 += __shfl_xor(s2, off, 32);
  }
  float mean = s * (1.0f / HDIM);
  float var  = s2 * (1.0f / HDIM) - mean * mean;
  float rstd = rsqrtf(var + LN_EPS);

  float* op = out + (long long)row * HDIM;
#pragma unroll
  for (int j = 0; j < 3; ++j) {
    int f = lane + 32 * j;
    float y = (v[j] - mean) * rstd * g[f] + b[f];
    if (acc) op[f] += y; else op[f] = y;
  }
}

// PE phi stage collapsed over K: s[n,f] = sum_k relu(zv*w1[0,f] + ze*w1[1,f])
__global__ void pe_s_kernel(const float* __restrict__ eigvecs,
                            const float* __restrict__ eigvals,
                            const float* __restrict__ pe_eps,
                            const float* __restrict__ phi_w1,
                            float* __restrict__ out, int n)
{
  int i = blockIdx.x * blockDim.x + threadIdx.x;
  if (i >= n * HDIM) return;
  int node = i / HDIM, f = i % HDIM;
  float w0 = phi_w1[f], w1 = phi_w1[HDIM + f];
  float s = 0.0f;
#pragma unroll
  for (int k = 0; k < KEIG; ++k) {
    float zv = fixnum(eigvecs[node * KEIG + k]);
    float ze = fixnum(eigvals[node * KEIG + k] + pe_eps[k]);
    s += fmaxf(zv * w0 + ze * w1, 0.0f);
  }
  out[i] = s;
}

__global__ void fill_kernel(float* __restrict__ p, int n, float v) {
  int i = blockIdx.x * blockDim.x + threadIdx.x;
  if (i < n) p[i] = v;
}

__global__ void add_kernel(float* __restrict__ dst, const float* __restrict__ src, int n) {
  int i = blockIdx.x * blockDim.x + threadIdx.x;
  if (i < n) dst[i] += src[i];
}

__global__ void deg_kernel(const int* __restrict__ dst, float* __restrict__ deg, int e) {
  int i = blockIdx.x * blockDim.x + threadIdx.x;
  if (i < e) atomAddF(&deg[dst[i]], 1.0f);
}

__global__ void dinv_kernel(const float* __restrict__ deg, float* __restrict__ dinv, int n) {
  int i = blockIdx.x * blockDim.x + threadIdx.x;
  if (i < n) dinv[i] = rsqrtf(deg[i] + 1.0f);  // +1 self-loop
}

// agg[dst,f] += xw[src,f] * dinv[src]*dinv[dst]; thread per (edge, feature)
__global__ void scatter_kernel(const float* __restrict__ xw,
                               const int* __restrict__ src,
                               const int* __restrict__ dst,
                               const float* __restrict__ dinv,
                               float* __restrict__ agg, int e)
{
  unsigned i = blockIdx.x * blockDim.x + threadIdx.x;
  unsigned total = (unsigned)e * HDIM;
  if (i >= total) return;
  int ei = i / HDIM, f = i % HDIM;
  int s = src[ei], d = dst[ei];
  float v = xw[(long long)s * HDIM + f] * dinv[s] * dinv[d];
  atomAddF(&agg[(long long)d * HDIM + f], v);
}

// Self-loop contribution + bias: agg += xw * dinv^2 + b
__global__ void selfloop_kernel(float* __restrict__ agg,
                                const float* __restrict__ xw,
                                const float* __restrict__ dinv,
                                const float* __restrict__ bias, int n)
{
  int i = blockIdx.x * blockDim.x + threadIdx.x;
  if (i >= n * HDIM) return;
  int node = i / HDIM, f = i % HDIM;
  float di = dinv[node];
  agg[i] += xw[i] * di * di + bias[f];
}

// Final tiny projection: out[n,c] = z[n,:96] @ head_w2[96,10] + b2
__global__ void head_kernel(const float* __restrict__ z,
                            const float* __restrict__ w2,
                            const float* __restrict__ b2,
                            float* __restrict__ out, int n)
{
  int i = blockIdx.x * blockDim.x + threadIdx.x;
  if (i >= n * CDIM) return;
  int node = i / CDIM, c = i % CDIM;
  float s = b2[c];
#pragma unroll 8
  for (int k = 0; k < HDIM; ++k)
    s += z[(long long)node * HDIM + k] * w2[k * CDIM + c];
  out[i] = s;
}

// ---------------------------------------------------------------------------
extern "C" void kernel_launch(void* const* d_in, const int* in_sizes, int n_in,
                              void* d_out, int out_size, void* d_ws, size_t ws_size,
                              hipStream_t stream)
{
  const float* x       = (const float*)d_in[0];
  const int*   edge    = (const int*)  d_in[1];
  const float* eigvecs = (const float*)d_in[2];
  const float* eigvals = (const float*)d_in[3];
  const float* enc_W   = (const float*)d_in[4];
  const float* enc_b   = (const float*)d_in[5];
  const float* in_g    = (const float*)d_in[6];
  const float* in_b    = (const float*)d_in[7];
  const float* phi_w1  = (const float*)d_in[8];
  const float* phi_w2  = (const float*)d_in[9];
  const float* rho_w1  = (const float*)d_in[10];
  const float* rho_b1  = (const float*)d_in[11];
  const float* rho_w2  = (const float*)d_in[12];
  const float* rho_b2  = (const float*)d_in[13];
  const float* pe_eps  = (const float*)d_in[14];
  const float* conv_W  = (const float*)d_in[15];
  const float* conv_b  = (const float*)d_in[16];
  const float* norm_g  = (const float*)d_in[17];
  const float* norm_b  = (const float*)d_in[18];
  const float* head_w1 = (const float*)d_in[19];
  const float* head_b1 = (const float*)d_in[20];
  const float* head_g  = (const float*)d_in[21];
  const float* head_b  = (const float*)d_in[22];
  const float* head_w2 = (const float*)d_in[23];
  const float* head_b2 = (const float*)d_in[24];
  const int* srcI = edge;
  const int* dstI = edge + NEDGES;

  float* ws  = (float*)d_ws;
  float* h   = ws;                                  // [N,96] residual stream
  float* t0  = h  + (size_t)NNODES * HDIM;          // [N,96] scratch
  float* t1  = t0 + (size_t)NNODES * HDIM;          // [N,96] scratch
  float* deg = t1 + (size_t)NNODES * HDIM;          // [N]
  float* dnv = deg + NNODES;                        // [N]
  _Float16* a16 = (_Float16*)(dnv + NNODES);        // [N,128] f16 A mirror

  const int NH  = NNODES * HDIM;
  const int NIN = NNODES * IN_DIM;
  int gEW   = (NH + 255) / 256;
  int gIN   = (NIN + 255) / 256;
  int gRows = (NNODES + 63) / 64;      // gemm blocks (64 rows each)
  int gLN   = (NNODES + 7) / 8;        // ln blocks (8 rows each)

  // --- Encoder: h = LN(x @ enc_W + enc_b) ---
  tof16_kernel<<<gIN, 256, 0, stream>>>(x, a16, NIN);
  gemm_wmma_kernel<<<gRows, 128, 0, stream>>>(a16, enc_W, enc_b, t0, NNODES, IN_DIM, 1, 0);
  ln_kernel<<<gLN, 256, 0, stream>>>(t0, in_g, in_b, h, NNODES, 0, 0);

  // --- PE encode: h += rho(relu-sum phi(z)) ---
  pe_s_kernel<<<gEW, 256, 0, stream>>>(eigvecs, eigvals, pe_eps, phi_w1, t0, NNODES);
  tof16_kernel<<<gEW, 256, 0, stream>>>(t0, a16, NH);
  gemm_wmma_kernel<<<gRows, 128, 0, stream>>>(a16, phi_w2, nullptr, t1, NNODES, HDIM, 0, 0);
  tof16_kernel<<<gEW, 256, 0, stream>>>(t1, a16, NH);
  gemm_wmma_kernel<<<gRows, 128, 0, stream>>>(a16, rho_w1, rho_b1, t0, NNODES, HDIM, 1, 1);
  tof16_kernel<<<gEW, 256, 0, stream>>>(t0, a16, NH);
  gemm_wmma_kernel<<<gRows, 128, 0, stream>>>(a16, rho_w2, rho_b2, t1, NNODES, HDIM, 1, 0);
  add_kernel<<<gEW, 256, 0, stream>>>(h, t1, NH);

  // --- Degrees (shared across layers) ---
  fill_kernel<<<(NNODES + 255) / 256, 256, 0, stream>>>(deg, NNODES, 0.0f);
  deg_kernel<<<(NEDGES + 255) / 256, 256, 0, stream>>>(dstI, deg, NEDGES);
  dinv_kernel<<<(NNODES + 255) / 256, 256, 0, stream>>>(deg, dnv, NNODES);

  // --- GCN layers ---
  unsigned scatterTotal = (unsigned)NEDGES * HDIM;
  int gScat = (int)((scatterTotal + 255u) / 256u);
  for (int l = 0; l < LAYERS; ++l) {
    tof16_kernel<<<gEW, 256, 0, stream>>>(h, a16, NH);
    gemm_wmma_kernel<<<gRows, 128, 0, stream>>>(a16, conv_W + (size_t)l * HDIM * HDIM,
                                                nullptr, t0, NNODES, HDIM, 0, 0);
    fill_kernel<<<gEW, 256, 0, stream>>>(t1, NH, 0.0f);
    scatter_kernel<<<gScat, 256, 0, stream>>>(t0, srcI, dstI, dnv, t1, NEDGES);
    selfloop_kernel<<<gEW, 256, 0, stream>>>(t1, t0, dnv, conv_b + l * HDIM, NNODES);
    // h += LN(gelu(agg)) * g + b
    ln_kernel<<<gLN, 256, 0, stream>>>(t1, norm_g + l * HDIM, norm_b + l * HDIM,
                                       h, NNODES, 1, 1);
  }

  // --- Head ---
  tof16_kernel<<<gEW, 256, 0, stream>>>(h, a16, NH);
  gemm_wmma_kernel<<<gRows, 128, 0, stream>>>(a16, head_w1, head_b1, t0, NNODES, HDIM, 1, 0);
  ln_kernel<<<gLN, 256, 0, stream>>>(t0, head_g, head_b, t1, NNODES, 1, 0);
  head_kernel<<<(NNODES * CDIM + 255) / 256, 256, 0, stream>>>(
      t1, head_w2, head_b2, (float*)d_out, NNODES);
}
